// SelfAttention_4449586119255
// MI455X (gfx1250) — compile-verified, matrix operand order checked by reading, not run on hardware
//
#include <hip/hip_runtime.h>

#define SEQ_L   2521
#define BATCH_N 2
#define NHEAD   16
#define HDIM    128
#define EMB     2048
#define MTOT    (BATCH_N * SEQ_L)   // 5042
#define NQT     20                  // ceil(2521/128)

typedef __attribute__((ext_vector_type(16))) _Float16 v16h;
typedef __attribute__((ext_vector_type(8)))  float    v8f;

union HV { v16h v; uint4 u[2]; };

__device__ __forceinline__ v8f v8f_zero() {
  v8f z = {0.f,0.f,0.f,0.f,0.f,0.f,0.f,0.f};
  return z;
}

// Async DMA: 16 bytes global -> LDS, per-lane addresses, ASYNCcnt-tracked.
// vdst = 32-bit LDS byte offset (low 32 bits of the generic shared pointer),
// vaddr = 64-bit global address, GV mode (saddr = off).
__device__ __forceinline__ void async_ld16(const void* g, const void* l) {
  asm volatile("global_load_async_to_lds_b128 %0, %1, off"
               :: "v"((unsigned)(unsigned long long)l),
                  "v"((unsigned long long)g)
               : "memory");
}
__device__ __forceinline__ void wait_async0() {
  asm volatile("s_wait_asynccnt 0" ::: "memory");
}

// VAR block-causal visibility: first cumulative scale length > l
__device__ __forceinline__ int q_end_of(int l) {
  const int cum[10] = {1, 5, 21, 57, 121, 265, 521, 921, 1497, 2521};
  int r = 2521;
#pragma unroll
  for (int i = 9; i >= 0; --i)
    if (l < cum[i]) r = cum[i];
  return r;
}

// ---------------------------------------------------------------- cast f32->f16
__global__ void cast_f2h(const float* __restrict__ s, _Float16* __restrict__ d, int n) {
  int stride = gridDim.x * blockDim.x;
  for (int i = blockIdx.x * blockDim.x + threadIdx.x; i < n; i += stride)
    d[i] = (_Float16)s[i];
}

// ---------------------------------------------------------------- WMMA GEMM
// out[M,N] = A[M,K] * B[N,K]^T (+bias).  MODE 0: QKV scatter to q/k/v f16.
// MODE 1: f32 out + proj bias.  Block tile 128x128, K-step 32, 8 waves,
// double-buffered ASYNCcnt pipeline: DMA tile t+1 overlaps WMMA of tile t.
template <int MODE>
__global__ __launch_bounds__(256) void gemm_wmma(
    const _Float16* __restrict__ Amat, const _Float16* __restrict__ Bmat,
    const float* __restrict__ qbias, const float* __restrict__ vbias,
    const float* __restrict__ pbias,
    _Float16* __restrict__ qh, _Float16* __restrict__ kh, _Float16* __restrict__ vh,
    float* __restrict__ outf, int K)
{
  __shared__ _Float16 smA[2 * 128 * 32];
  __shared__ _Float16 smB[2 * 128 * 32];
  const int tid  = threadIdx.x;
  const int lane = tid & 31;
  const int w    = tid >> 5;
  const int wm   = w & 1, wn = w >> 1;
  const int m0   = blockIdx.x * 128;
  const int n0   = blockIdx.y * 128;
  const int hi   = lane >> 4;

  v8f acc[4][2];
#pragma unroll
  for (int i = 0; i < 4; i++)
#pragma unroll
    for (int j = 0; j < 2; j++) acc[i][j] = v8f_zero();

  // stage tile kt into LDS buffer bufsel (fragment-swizzled, 16B async chunks)
  auto stage = [&](int kt, int bufsel) {
    int k0 = kt << 5;
#pragma unroll
    for (int it = 0; it < 2; ++it) {
      int c   = tid + it * 256;          // 0..511 chunks of 8 halves
      int row = c >> 2, kc = c & 3;
      int loff = bufsel * 4096 +
                 (row >> 4) * 512 + ((row & 15) + 16 * (kc & 1)) * 16 + ((kc >> 1) & 1) * 8;
      int gm = m0 + row; if (gm >= MTOT) gm = MTOT - 1;   // clamp: rows unused in epilogue
      async_ld16(Amat + (size_t)gm * K + k0 + kc * 8, smA + loff);
      int gn = n0 + row;
      async_ld16(Bmat + (size_t)gn * K + k0 + kc * 8, smB + loff);
    }
  };

  const int KT = K >> 5;
  stage(0, 0);
  for (int kt = 0; kt < KT; ++kt) {
    int cur = kt & 1;
    wait_async0();        // my DMA for tile kt has landed
    __syncthreads();      // everyone's DMA landed; everyone done reading other buf
    if (kt + 1 < KT) stage(kt + 1, cur ^ 1);   // overlaps WMMAs below

    v16h af[4], bf[2];
#pragma unroll
    for (int i = 0; i < 4; i++)
      af[i] = *reinterpret_cast<const v16h*>(smA + cur * 4096 + (wm * 4 + i) * 512 + lane * 16);
#pragma unroll
    for (int j = 0; j < 2; j++)
      bf[j] = *reinterpret_cast<const v16h*>(smB + cur * 4096 + (wn * 2 + j) * 512 + lane * 16);
#pragma unroll
    for (int i = 0; i < 4; i++)
#pragma unroll
      for (int j = 0; j < 2; j++)
        acc[i][j] = __builtin_amdgcn_wmma_f32_16x16x32_f16(
            false, af[i], false, bf[j], (short)0, acc[i][j], false, false);
  }

  // epilogue
#pragma unroll
  for (int i = 0; i < 4; i++) {
#pragma unroll
    for (int j = 0; j < 2; j++) {
      int gn = n0 + (wn * 2 + j) * 16 + (lane & 15);
#pragma unroll
      for (int r = 0; r < 8; r++) {
        int gm = m0 + (wm * 4 + i) * 16 + r + 8 * hi;
        if (gm < MTOT) {
          float v = acc[i][j][r];
          if (MODE == 0) {
            int which = gn >> 11;          // 0:q 1:k 2:v
            int col   = gn & 2047;
            float bias = (which == 0) ? qbias[col] : ((which == 2) ? vbias[col] : 0.f);
            v += bias;
            int head = col >> 7, d = col & 127;
            int b = gm / SEQ_L, l = gm - b * SEQ_L;
            _Float16* dst = (which == 0) ? qh : ((which == 1) ? kh : vh);
            dst[(((size_t)b * NHEAD + head) * SEQ_L + l) * HDIM + d] = (_Float16)v;
          } else {
            outf[(size_t)gm * EMB + gn] = v + pbias[gn];
          }
        }
      }
    }
  }
}

// ---------------------------------------------------------------- l2norm + cos-scale + RoPE
__global__ __launch_bounds__(256) void norm_rope(
    _Float16* __restrict__ qh, _Float16* __restrict__ kh,
    const float* __restrict__ smul_log, const float* __restrict__ rope)
{
  int row  = blockIdx.x * 8 + (threadIdx.x >> 5);   // grid covers B*H*L exactly
  int lane = threadIdx.x & 31;
  int hl = row % (NHEAD * SEQ_L);
  int h  = hl / SEQ_L;
  int l  = hl - h * SEQ_L;
  _Float16* qp = qh + (size_t)row * HDIM + lane * 4;
  _Float16* kp = kh + (size_t)row * HDIM + lane * 4;
  float qv[4], kv[4];
#pragma unroll
  for (int j = 0; j < 4; j++) { qv[j] = (float)qp[j]; kv[j] = (float)kp[j]; }
  float sq = qv[0]*qv[0] + qv[1]*qv[1] + qv[2]*qv[2] + qv[3]*qv[3];
  float sk = kv[0]*kv[0] + kv[1]*kv[1] + kv[2]*kv[2] + kv[3]*kv[3];
#pragma unroll
  for (int m = 16; m >= 1; m >>= 1) {
    sq += __shfl_xor(sq, m, 32);
    sk += __shfl_xor(sk, m, 32);
  }
  float smul = __expf(fminf(smul_log[h], 4.6051702f));   // min(log-scale, log 100)
  float qs = smul / fmaxf(sqrtf(sq), 1e-12f);
  float ks = 1.0f / fmaxf(sqrtf(sk), 1e-12f);
  const float* cosp = rope + (size_t)l * (HDIM / 2);
  const float* sinp = rope + (size_t)SEQ_L * (HDIM / 2) + (size_t)l * (HDIM / 2);
#pragma unroll
  for (int p = 0; p < 2; p++) {
    float cc = cosp[lane * 2 + p], ss = sinp[lane * 2 + p];
    float q0 = qv[2*p] * qs, q1 = qv[2*p+1] * qs;
    qp[2*p]   = (_Float16)(cc * q0 - ss * q1);
    qp[2*p+1] = (_Float16)(ss * q0 + cc * q1);
    float k0 = kv[2*p] * ks, k1 = kv[2*p+1] * ks;
    kp[2*p]   = (_Float16)(cc * k0 - ss * k1);
    kp[2*p+1] = (_Float16)(ss * k0 + cc * k1);
  }
}

// ---------------------------------------------------------------- flash attention
// block = (b, h, 128-query tile); wave w owns rows [q0+16w, q0+16w+16), full hd.
__global__ __launch_bounds__(256) void flash_attn(
    const _Float16* __restrict__ qg, const _Float16* __restrict__ kg,
    const _Float16* __restrict__ vg, _Float16* __restrict__ aout)
{
  __shared__ _Float16 smK[4096];   // 32 kv x 128 hd, B-frag swizzled (async DMA)
  __shared__ _Float16 smV[4096];   // 32 kv x 128 d, transposed to B-frag layout
  __shared__ _Float16 smP[4096];   // per-wave 16x32 P tiles in A-frag layout
  const int tid = threadIdx.x, lane = tid & 31, w = tid >> 5;
  const int hi = lane >> 4;
  int blk = blockIdx.x;
  int qt  = blk % NQT;
  int bh  = blk / NQT;
  int h   = bh & 15, b = bh >> 4;
  int q0  = qt * 128;
  int wq0 = q0 + w * 16;

  // persistent Q A-fragments for 4 K-chunks of hd
  int qrow = wq0 + (lane & 15);
  if (qrow >= SEQ_L) qrow = SEQ_L - 1;
  const _Float16* qptr = qg + ((size_t)bh * SEQ_L + qrow) * HDIM;
  v16h qf[4];
#pragma unroll
  for (int c = 0; c < 4; c++) {
    HV t;
    t.u[0] = *reinterpret_cast<const uint4*>(qptr + c * 32 + hi * 8);
    t.u[1] = *reinterpret_cast<const uint4*>(qptr + c * 32 + 16 + hi * 8);
    qf[c] = t.v;
  }
  int qend[8];
#pragma unroll
  for (int r = 0; r < 8; r++) {
    int rr = wq0 + r + 8 * hi;
    qend[r] = (rr < SEQ_L) ? q_end_of(rr) : 0;
  }
  float mrow[8], lrow[8];
  v8f o[8];
#pragma unroll
  for (int r = 0; r < 8; r++) { mrow[r] = -1e30f; lrow[r] = 0.f; }
#pragma unroll
  for (int d = 0; d < 8; d++) o[d] = v8f_zero();

  int qlast = q0 + 127; if (qlast >= SEQ_L) qlast = SEQ_L - 1;
  int ntiles = (q_end_of(qlast) + 31) >> 5;   // block-uniform trip count

  for (int t = 0; t < ntiles; ++t) {
    int t0 = t << 5;
    // stage K via async DMA (clamped rows get masked by q_end select below),
    // V via transpose scatter (2-byte granularity).
#pragma unroll
    for (int it = 0; it < 2; ++it) {
      int c = tid + it * 256;              // 512 chunks of 8 halves
      int kv = c >> 4, dc = c & 15;
      int kvg = t0 + kv;
      int kvc = (kvg < SEQ_L) ? kvg : (SEQ_L - 1);
      async_ld16(kg + ((size_t)bh * SEQ_L + kvc) * HDIM + dc * 8,
                 smK + (dc >> 2) * 1024 + (kv >> 4) * 512 +
                     ((kv & 15) + 16 * (dc & 1)) * 16 + ((dc >> 1) & 1) * 8);
      uint4 vv; vv.x = vv.y = vv.z = vv.w = 0u;
      if (kvg < SEQ_L)
        vv = *reinterpret_cast<const uint4*>(vg + ((size_t)bh * SEQ_L + kvg) * HDIM + dc * 8);
      HV hv; hv.u[0] = vv; hv.u[1] = vv;
      int hh = (kv & 7) + 8 * (kv >> 4);
#pragma unroll
      for (int j = 0; j < 8; j++) {
        int d = dc * 8 + j;
        smV[(d >> 4) * 512 + ((d & 15) + 16 * ((kv >> 3) & 1)) * 16 + hh] = hv.v[j];
      }
    }
    wait_async0();
    __syncthreads();

    // S = Q * K^T  (16 x 32 per wave); load all frags of an n-tile, then WMMA
    v8f S[2]; S[0] = v8f_zero(); S[1] = v8f_zero();
#pragma unroll
    for (int nt = 0; nt < 2; ++nt) {
      v16h bfa[4];
#pragma unroll
      for (int c = 0; c < 4; c++)
        bfa[c] = *reinterpret_cast<const v16h*>(smK + c * 1024 + nt * 512 + lane * 16);
#pragma unroll
      for (int c = 0; c < 4; c++)
        S[nt] = __builtin_amdgcn_wmma_f32_16x16x32_f16(
            false, qf[c], false, bfa[c], (short)0, S[nt], false, false);
    }

    // block-causal mask (select, keeps EXEC all-1s) + online softmax
    float sv[2][8], tmax[8], tsum[8], resc[8];
#pragma unroll
    for (int r = 0; r < 8; r++) tmax[r] = -1e30f;
#pragma unroll
    for (int nt = 0; nt < 2; ++nt)
#pragma unroll
      for (int r = 0; r < 8; r++) {
        int col = t0 + nt * 16 + (lane & 15);
        float x = S[nt][r];
        x = (col < qend[r]) ? x : -1e30f;
        sv[nt][r] = x;
        tmax[r] = fmaxf(tmax[r], x);
      }
#pragma unroll
    for (int r = 0; r < 8; r++) {
#pragma unroll
      for (int m = 8; m >= 1; m >>= 1)
        tmax[r] = fmaxf(tmax[r], __shfl_xor(tmax[r], m, 32));
      float nm = fmaxf(mrow[r], tmax[r]);
      resc[r] = __expf(mrow[r] - nm);
      mrow[r] = nm;
      tsum[r] = 0.f;
    }
#pragma unroll
    for (int nt = 0; nt < 2; ++nt)
#pragma unroll
      for (int r = 0; r < 8; r++) {
        float p = __expf(sv[nt][r] - mrow[r]);
        sv[nt][r] = p;
        tsum[r] += p;
      }
#pragma unroll
    for (int r = 0; r < 8; r++) {
#pragma unroll
      for (int m = 8; m >= 1; m >>= 1)
        tsum[r] += __shfl_xor(tsum[r], m, 32);
      lrow[r] = lrow[r] * resc[r] + tsum[r];
    }
#pragma unroll
    for (int d = 0; d < 8; d++)
#pragma unroll
      for (int r = 0; r < 8; r++) o[d][r] *= resc[r];

    // P (C-layout) -> per-wave LDS in A-fragment layout, then O += P*V
#pragma unroll
    for (int nt = 0; nt < 2; ++nt)
#pragma unroll
      for (int r = 0; r < 8; r++) {
        int mr = r + 8 * hi;
        int n  = nt * 16 + (lane & 15);
        smP[w * 512 + (mr + 16 * ((n >> 3) & 1)) * 16 + (n & 7) + 8 * (n >> 4)] =
            (_Float16)sv[nt][r];
      }
    asm volatile("s_wait_dscnt 0" ::: "memory");   // same-wave LDS RAW
    v16h pf = *reinterpret_cast<const v16h*>(smP + w * 512 + lane * 16);
#pragma unroll
    for (int d = 0; d < 8; d++) {
      v16h vf = *reinterpret_cast<const v16h*>(smV + d * 512 + lane * 16);
      o[d] = __builtin_amdgcn_wmma_f32_16x16x32_f16(
          false, pf, false, vf, (short)0, o[d], false, false);
    }
    __syncthreads();
  }

  // normalize, write to [B,L,C] f16 for the projection GEMM
#pragma unroll
  for (int d = 0; d < 8; d++)
#pragma unroll
    for (int r = 0; r < 8; r++) {
      int qr = wq0 + r + 8 * hi;
      if (qr < SEQ_L) {
        float v = o[d][r] / fmaxf(lrow[r], 1e-20f);
        aout[((size_t)b * SEQ_L + qr) * EMB + h * HDIM + d * 16 + (lane & 15)] = (_Float16)v;
      }
    }
}

// ---------------------------------------------------------------- launch
extern "C" void kernel_launch(void* const* d_in, const int* in_sizes, int n_in,
                              void* d_out, int out_size, void* d_ws, size_t ws_size,
                              hipStream_t stream) {
  (void)in_sizes; (void)n_in; (void)out_size; (void)ws_size;
  const float* x    = (const float*)d_in[0];
  const float* Wqkv = (const float*)d_in[1];
  const float* qb   = (const float*)d_in[2];
  const float* vb   = (const float*)d_in[3];
  const float* sml  = (const float*)d_in[4];
  const float* pW   = (const float*)d_in[5];
  const float* pb   = (const float*)d_in[6];
  const float* rope = (const float*)d_in[7];
  // d_in[8] = attn_bias (recomputed analytically on device)

  char* ws = (char*)d_ws;
  size_t off = 0;
  auto take = [&](size_t halves) -> _Float16* {
    _Float16* p = (_Float16*)(ws + off);
    off += (halves * sizeof(_Float16) + 255) & ~(size_t)255;
    return p;
  };
  const size_t nX = (size_t)MTOT * EMB;      // 10,326,016
  const size_t nW = (size_t)3 * EMB * EMB;   // 12,582,912
  const size_t nP = (size_t)EMB * EMB;
  _Float16* xh  = take(nX);
  _Float16* wh  = take(nW);
  _Float16* pwh = take(nP);
  _Float16* qh  = take(nX);
  _Float16* kh  = take(nX);
  _Float16* vh  = take(nX);
  _Float16* ah  = take(nX);

  cast_f2h<<<2048, 256, 0, stream>>>(x, xh, (int)nX);
  cast_f2h<<<2048, 256, 0, stream>>>(Wqkv, wh, (int)nW);
  cast_f2h<<<1024, 256, 0, stream>>>(pW, pwh, (int)nP);

  gemm_wmma<0><<<dim3(40, 48), 256, 0, stream>>>(
      xh, wh, qb, vb, nullptr, qh, kh, vh, nullptr, EMB);
  norm_rope<<<(BATCH_N * NHEAD * SEQ_L) / 8, 256, 0, stream>>>(qh, kh, sml, rope);
  flash_attn<<<BATCH_N * NHEAD * NQT, 256, 0, stream>>>(qh, kh, vh, ah);
  gemm_wmma<1><<<dim3(40, 16), 256, 0, stream>>>(
      ah, pwh, nullptr, nullptr, pb, nullptr, nullptr, nullptr, (float*)d_out, EMB);
}
